// GRUMasked_39573828666300
// MI455X (gfx1250) — matrix operand order repaired and checked by reading, not compiled, not run on hardware
//
#include <hip/hip_runtime.h>
#include <hip/hip_bf16.h>
#include <math.h>

// Problem dims (match reference)
#define BB 128
#define SS 256
#define II 512
#define HH 512
#define G3 1536   // 3*H

typedef __attribute__((ext_vector_type(16))) __bf16 v16bf;
typedef __attribute__((ext_vector_type(8)))  float  v8f;

__device__ __forceinline__ float sigmoidf_(float x) {
  return 1.0f / (1.0f + __expf(-x));
}

// A-fragment K order within a 32-wide K block: contiguous (identity),
// matching the contiguous B-fragment loads.
__device__ __forceinline__ int permK(int k) {
  int g = (k >> 3) & 3;
  int pg = ((g << 1) | (g >> 1)) & 3;
  return k;
}

#define WMMA_BF16(A, B, C) \
  __builtin_amdgcn_wmma_f32_16x16x32_bf16(false, (A), false, (B), (short)0, (C), false, false)

// ---------------------------------------------------------------------------
// Kernel 0a: split weights into bf16 hi/lo planes; init masked h0 (f32 +
// A-fragment-ordered bf16 hi/lo); zero barrier counter.
// ---------------------------------------------------------------------------
__global__ __launch_bounds__(256) void gru_prep_w_kernel(
    const float* __restrict__ Wih, const float* __restrict__ Whh,
    const float* __restrict__ hx, const float* __restrict__ mask,
    unsigned short* __restrict__ WihHi, unsigned short* __restrict__ WihLo,
    unsigned short* __restrict__ WhhHi, unsigned short* __restrict__ WhhLo,
    float* __restrict__ hmF, unsigned short* __restrict__ hmHi,
    unsigned short* __restrict__ hmLo, unsigned* __restrict__ cnt) {
  int i = blockIdx.x * blockDim.x + threadIdx.x;
  if (i < G3 * II) {
    float a = Wih[i];
    __bf16 ah = (__bf16)a;
    WihHi[i] = __builtin_bit_cast(unsigned short, ah);
    __bf16 al = (__bf16)(a - (float)ah);
    WihLo[i] = __builtin_bit_cast(unsigned short, al);

    float b = Whh[i];
    __bf16 bh = (__bf16)b;
    WhhHi[i] = __builtin_bit_cast(unsigned short, bh);
    __bf16 bl = (__bf16)(b - (float)bh);
    WhhLo[i] = __builtin_bit_cast(unsigned short, bl);
  }
  if (i < BB * HH) {
    int b = i >> 9, k = i & 511;
    float v = hx[i] * mask[b * SS];     // masked h for step 0
    hmF[i] = v;
    int pos = b * HH + permK(k);
    __bf16 h = (__bf16)v;
    hmHi[pos] = __builtin_bit_cast(unsigned short, h);
    __bf16 l = (__bf16)(v - (float)h);
    hmLo[pos] = __builtin_bit_cast(unsigned short, l);
  }
  if (i == 0) *cnt = 0u;
}

// ---------------------------------------------------------------------------
// Kernel 0b: split x into bf16 hi/lo, A-fragment K order (done ONCE instead
// of 96x redundantly inside the gi GEMM).
// ---------------------------------------------------------------------------
__global__ __launch_bounds__(256) void gru_prep_x_kernel(
    const float* __restrict__ x,
    unsigned short* __restrict__ xHi, unsigned short* __restrict__ xLo) {
  int i = blockIdx.x * blockDim.x + threadIdx.x;  // over B*S*I = 16.7M
  int row = i >> 9, k = i & 511;
  float a = x[i];
  int pos = row * II + permK(k);
  __bf16 h = (__bf16)a;
  xHi[pos] = __builtin_bit_cast(unsigned short, h);
  __bf16 l = (__bf16)(a - (float)h);
  xLo[pos] = __builtin_bit_cast(unsigned short, l);
}

// ---------------------------------------------------------------------------
// Kernel 1: gi = x @ W_ih^T + b_ih   ([B*S, 3H], K=I), bf16x3, pure load+WMMA.
// Waves in a block share the N tile -> B fragments hit WGP$.
// ---------------------------------------------------------------------------
__global__ __launch_bounds__(256) void gru_gi_kernel(
    const unsigned short* __restrict__ xHiU, const unsigned short* __restrict__ xLoU,
    const unsigned short* __restrict__ WHiU, const unsigned short* __restrict__ WLoU,
    const float* __restrict__ b_ih, float* __restrict__ gi) {
  const __bf16* xHi = reinterpret_cast<const __bf16*>(xHiU);
  const __bf16* xLo = reinterpret_cast<const __bf16*>(xLoU);
  const __bf16* WHi = reinterpret_cast<const __bf16*>(WHiU);
  const __bf16* WLo = reinterpret_cast<const __bf16*>(WLoU);

  const int lane = threadIdx.x & 31;
  const int wv   = blockIdx.x * (blockDim.x >> 5) + (threadIdx.x >> 5);
  const int tn   = wv >> 11;          // 0..95  (same within a block)
  const int tm   = wv & 2047;         // 0..2047 row tiles over B*S
  const int m0   = tm * 16, n0 = tn * 16;
  const int l15  = lane & 15, h16 = lane >> 4;

  const __bf16* aHi = xHi + (size_t)(m0 + l15) * II;
  const __bf16* aLo = xLo + (size_t)(m0 + l15) * II;
  const size_t bRow = (size_t)(n0 + l15) * II;

  v8f acc0 = {}, acc1 = {};
  for (int k0 = 0; k0 < II; k0 += 32) {
    const int kb = k0 + 16 * h16;
    v16bf ahi = *(const v16bf*)(aHi + kb);
    v16bf alo = *(const v16bf*)(aLo + kb);
    v16bf bhi = *(const v16bf*)(WHi + bRow + kb);
    v16bf blo = *(const v16bf*)(WLo + bRow + kb);
    acc0 = WMMA_BF16(ahi, bhi, acc0);
    acc1 = WMMA_BF16(alo, bhi, acc1);
    acc1 = WMMA_BF16(ahi, blo, acc1);
  }
  v8f acc = acc0 + acc1;

  const int col  = n0 + l15;
  const float bias = b_ih[col];
#pragma unroll
  for (int v = 0; v < 8; ++v) {
    int row = m0 + v + 8 * h16;
    gi[(size_t)row * G3 + col] = acc[v] + bias;
  }
}

// ---------------------------------------------------------------------------
// Kernel 2: persistent masked-GRU scan, K-split 4-ways across waves.
// 128 blocks x 8 waves. Block = (tn, batch-tile-pair); wave = (tm_local, ks).
// ks>0 partials reduced via LDS; ks==0 waves run the gate epilogue.
// Epilogue stores h_{t+1}*mask pre-masked AND pre-split to bf16 hi/lo in
// A-fragment order, so the inner loop is pure load+WMMA.
// ---------------------------------------------------------------------------
__global__ __launch_bounds__(256) void gru_scan_kernel(
    const float* __restrict__ mask, const float* __restrict__ b_hh,
    const unsigned short* __restrict__ WHiU, const unsigned short* __restrict__ WLoU,
    const float* __restrict__ gi,
    float* __restrict__ hmF, unsigned short* __restrict__ hmHiU,
    unsigned short* __restrict__ hmLoU,
    float* __restrict__ out, unsigned* __restrict__ cnt) {
  const __bf16* WHi = reinterpret_cast<const __bf16*>(WHiU);
  const __bf16* WLo = reinterpret_cast<const __bf16*>(WLoU);

  // Partial-sum exchange: [slot 0..5][j 0..23][lane 0..31], lane-contiguous
  // -> conflict-free LDS banks.
  __shared__ float red[6 * 24 * 32];   // 18 KB

  const int lane = threadIdx.x & 31;
  const int w    = threadIdx.x >> 5;   // 0..7
  const int tn   = blockIdx.x >> 2;    // 0..31 hidden-column tile (shared in block)
  const int tmg  = blockIdx.x & 3;     // batch-tile pair index
  const int tml  = w & 1;              // 0..1
  const int ks   = w >> 1;             // 0..3 K quarter
  const int tm   = tmg * 2 + tml;      // 0..7 batch tile
  const int m0   = tm * 16, n0 = tn * 16;
  const int l15  = lane & 15, h16 = lane >> 4;
  const int kstart = ks * (HH / 4);    // 0,128,256,384

  const int col = n0 + l15;
  const float bhr = b_hh[col];
  const float bhz = b_hh[HH + col];
  const float bhn = b_hh[2 * HH + col];
  const int permCol = permK(col);      // epilogue store position for this lane's col

  const size_t wr = (size_t)(0 * HH + col) * HH;
  const size_t wz = (size_t)(1 * HH + col) * HH;
  const size_t wn = (size_t)(2 * HH + col) * HH;

  for (int t = 0; t < SS; ++t) {
    const int p = t & 1;
    const float*  hmFp = hmF + (size_t)p * (BB * HH);
    float*        hmFn = hmF + (size_t)(p ^ 1) * (BB * HH);
    const __bf16* hmHi = reinterpret_cast<const __bf16*>(hmHiU + (size_t)p * (BB * HH));
    const __bf16* hmLo = reinterpret_cast<const __bf16*>(hmLoU + (size_t)p * (BB * HH));
    unsigned short* hmHiN = hmHiU + (size_t)(p ^ 1) * (BB * HH);
    unsigned short* hmLoN = hmLoU + (size_t)(p ^ 1) * (BB * HH);

    const int brow = m0 + l15;
    const __bf16* aHi = hmHi + (size_t)brow * HH;
    const __bf16* aLo = hmLo + (size_t)brow * HH;

    // 6 independent accumulator chains (hi-product + lo-products per gate)
    v8f r0 = {}, r1 = {}, z0 = {}, z1 = {}, n0v = {}, n1v = {};
#pragma unroll
    for (int k0 = kstart; k0 < kstart + HH / 4; k0 += 32) {
      const int kb = k0 + 16 * h16;
      v16bf ahi = *(const v16bf*)(aHi + kb);
      v16bf alo = *(const v16bf*)(aLo + kb);
      {
        v16bf bh = *(const v16bf*)(WHi + wr + kb);
        v16bf bl = *(const v16bf*)(WLo + wr + kb);
        r0 = WMMA_BF16(ahi, bh, r0);
        r1 = WMMA_BF16(alo, bh, r1);
        r1 = WMMA_BF16(ahi, bl, r1);
      }
      {
        v16bf bh = *(const v16bf*)(WHi + wz + kb);
        v16bf bl = *(const v16bf*)(WLo + wz + kb);
        z0 = WMMA_BF16(ahi, bh, z0);
        z1 = WMMA_BF16(alo, bh, z1);
        z1 = WMMA_BF16(ahi, bl, z1);
      }
      {
        v16bf bh = *(const v16bf*)(WHi + wn + kb);
        v16bf bl = *(const v16bf*)(WLo + wn + kb);
        n0v = WMMA_BF16(ahi, bh, n0v);
        n1v = WMMA_BF16(alo, bh, n1v);
        n1v = WMMA_BF16(ahi, bl, n1v);
      }
    }
    v8f accr = r0 + r1;
    v8f accz = z0 + z1;
    v8f accn = n0v + n1v;

    // K-split reduction through LDS.
    if (ks != 0) {
      const int slot = w - 2;   // 0..5
#pragma unroll
      for (int j = 0; j < 8; ++j) {
        red[((slot * 24 + j) * 32) + lane]      = accr[j];
        red[((slot * 24 + 8 + j) * 32) + lane]  = accz[j];
        red[((slot * 24 + 16 + j) * 32) + lane] = accn[j];
      }
    }
    __syncthreads();
    if (ks == 0) {
#pragma unroll
      for (int s = 0; s < 3; ++s) {
        const int slot = 2 * s + tml;
#pragma unroll
        for (int j = 0; j < 8; ++j) {
          accr[j] += red[((slot * 24 + j) * 32) + lane];
          accz[j] += red[((slot * 24 + 8 + j) * 32) + lane];
          accn[j] += red[((slot * 24 + 16 + j) * 32) + lane];
        }
      }

      // Gate math + stores (C layout: element v -> row m0+v+8*h16, col n0+l15).
#pragma unroll
      for (int v = 0; v < 8; ++v) {
        const int b2 = m0 + v + 8 * h16;
        const float hm2 = hmFp[(size_t)b2 * HH + col];   // already masked h_t
        const size_t gib = ((size_t)b2 * SS + t) * G3;
        const float ir  = gi[gib + col];
        const float iz  = gi[gib + HH + col];
        const float inn = gi[gib + 2 * HH + col];
        const float r = sigmoidf_(ir + accr[v] + bhr);
        const float z = sigmoidf_(iz + accz[v] + bhz);
        const float n = tanhf(inn + r * (accn[v] + bhn));
        const float hnew = (1.0f - z) * n + z * hm2;
        out[((size_t)b2 * SS + t) * HH + col] = hnew;
        if (t + 1 < SS) {
          // Pre-mask and pre-split next step's hidden state.
          const float hmn = hnew * mask[b2 * SS + (t + 1)];
          hmFn[(size_t)b2 * HH + col] = hmn;
          __bf16 hh = (__bf16)hmn;
          hmHiN[(size_t)b2 * HH + permCol] = __builtin_bit_cast(unsigned short, hh);
          __bf16 ll = (__bf16)(hmn - (float)hh);
          hmLoN[(size_t)b2 * HH + permCol] = __builtin_bit_cast(unsigned short, ll);
        } else {
          out[(size_t)BB * SS * HH + (size_t)b2 * HH + col] = hnew;  // h_last
        }
      }
    }

    // ---- grid-wide barrier (all 128 blocks resident). The two syncthreads
    // here also fence LDS reuse across steps. ----
    __threadfence();
    __syncthreads();
    if (threadIdx.x == 0) {
      __hip_atomic_fetch_add(cnt, 1u, __ATOMIC_RELEASE, __HIP_MEMORY_SCOPE_AGENT);
      const unsigned target = (unsigned)gridDim.x * (unsigned)(t + 1);
      while (__hip_atomic_load(cnt, __ATOMIC_ACQUIRE, __HIP_MEMORY_SCOPE_AGENT) < target) {
        __builtin_amdgcn_s_sleep(2);
      }
    }
    __syncthreads();
    __threadfence();
  }
}

// ---------------------------------------------------------------------------
// Host launcher
// ---------------------------------------------------------------------------
extern "C" void kernel_launch(void* const* d_in, const int* in_sizes, int n_in,
                              void* d_out, int out_size, void* d_ws, size_t ws_size,
                              hipStream_t stream) {
  const float* x    = (const float*)d_in[0];
  const float* hx   = (const float*)d_in[1];
  const float* mask = (const float*)d_in[2];
  const float* W_ih = (const float*)d_in[3];
  const float* W_hh = (const float*)d_in[4];
  const float* b_ih = (const float*)d_in[5];
  const float* b_hh = (const float*)d_in[6];
  float* out = (float*)d_out;

  // Workspace layout (all chunks 32B-aligned by construction)
  char* ws = (char*)d_ws;
  size_t off = 0;
  float* gi = (float*)(ws + off);                       off += (size_t)BB * SS * G3 * 4;  // 201 MB
  unsigned short* xHi  = (unsigned short*)(ws + off);   off += (size_t)BB * SS * II * 2;  // 33.5 MB
  unsigned short* xLo  = (unsigned short*)(ws + off);   off += (size_t)BB * SS * II * 2;
  unsigned short* WihHi = (unsigned short*)(ws + off);  off += (size_t)G3 * II * 2;
  unsigned short* WihLo = (unsigned short*)(ws + off);  off += (size_t)G3 * II * 2;
  unsigned short* WhhHi = (unsigned short*)(ws + off);  off += (size_t)G3 * HH * 2;
  unsigned short* WhhLo = (unsigned short*)(ws + off);  off += (size_t)G3 * HH * 2;
  float* hmF = (float*)(ws + off);                      off += (size_t)2 * BB * HH * 4;
  unsigned short* hmHi = (unsigned short*)(ws + off);   off += (size_t)2 * BB * HH * 2;
  unsigned short* hmLo = (unsigned short*)(ws + off);   off += (size_t)2 * BB * HH * 2;
  unsigned* cnt = (unsigned*)(ws + off);                off += 64;
  if (off > ws_size) return;  // workspace too small: deterministic no-op

  // Kernel 0a: weight split + masked h0 + barrier reset
  gru_prep_w_kernel<<<(G3 * II + 255) / 256, 256, 0, stream>>>(
      W_ih, W_hh, hx, mask, WihHi, WihLo, WhhHi, WhhLo, hmF, hmHi, hmLo, cnt);

  // Kernel 0b: x split (B*S*I threads)
  gru_prep_x_kernel<<<(BB * SS * II) / 256, 256, 0, stream>>>(x, xHi, xLo);

  // Kernel 1: gi GEMM (2048 x 96 tiles, 8 waves/block sharing the N tile)
  {
    int tiles = (BB * SS / 16) * (G3 / 16);   // 196608
    gru_gi_kernel<<<tiles / 8, 256, 0, stream>>>(xHi, xLo, WihHi, WihLo, b_ih, gi);
  }

  // Kernel 2: persistent K-split scan (128 blocks x 256 threads)
  gru_scan_kernel<<<128, 256, 0, stream>>>(mask, b_hh, WhhHi, WhhLo, gi,
                                           hmF, hmHi, hmLo, out, cnt);
}
